// GraphAttentionNetwork_66391604461926
// MI455X (gfx1250) — matrix operand order repaired and checked by reading, not compiled, not run on hardware
//
#include <hip/hip_runtime.h>
#include <hip/hip_bf16.h>
#include <math.h>

// ---------------- problem constants ----------------
#define NNODES 50000
#define NEDGES 800000
#define FIN    128
#define HIDD   64
#define NHEADS 4
#define NCLS   16
#define NEG_SLOPE 0.2f
#define HD1 (NHEADS * HIDD)   // 256

typedef __attribute__((ext_vector_type(16))) _Float16 v16h;
typedef __attribute__((ext_vector_type(8)))  _Float16 h8v;
typedef __attribute__((ext_vector_type(8)))  float    v8f;

// ---------------- helpers ----------------
__device__ __forceinline__ void atomicMaxF32(float* addr, float val) {
  // ordered-float max via integer atomics; valid with -inf init
  if (val >= 0.0f) atomicMax((int*)addr, __float_as_int(val));
  else             atomicMin((unsigned int*)addr, __float_as_uint(val));
}

// ---------------- conversion / fill kernels ----------------
__global__ void k_f32_to_f16(const float* __restrict__ src,
                             _Float16* __restrict__ dst, int n) {
  int i = blockIdx.x * blockDim.x + threadIdx.x;
  if (i < n) dst[i] = (_Float16)src[i];
}

// W[K,Nout] (row-major) -> WT[Nout,K] f16
__global__ void k_transpose_f16(const float* __restrict__ W,
                                _Float16* __restrict__ WT, int K, int Nout) {
  int i = blockIdx.x * blockDim.x + threadIdx.x;
  if (i < K * Nout) {
    int k = i / Nout, n = i % Nout;
    WT[(size_t)n * K + k] = (_Float16)W[i];
  }
}

__global__ void k_fill(float* __restrict__ p, int n, float v) {
  int i = blockIdx.x * blockDim.x + threadIdx.x;
  if (i < n) p[i] = v;
}

// ---------------- WMMA GEMM: C[M,Nout] = A[M,K](f16) * BT[Nout,K](f16)^T ----------------
__global__ void __launch_bounds__(128)
k_gemm_wmma(const _Float16* __restrict__ A, const _Float16* __restrict__ BT,
            float* __restrict__ C, int M, int K, int Nout) {
  const int lane  = threadIdx.x;                       // 0..31, wave32
  const int tileN = blockIdx.y * blockDim.y + threadIdx.y;
  if (tileN * 16 >= Nout) return;                      // wave-uniform
  const int row0 = blockIdx.x * 16;
  const int col0 = tileN * 16;

  // A fragment: lanes 0-15 -> M=lane, K base 0; lanes 16-31 -> M=lane-16, K base 8
  const int mrow  = row0 + (lane & 15);
  const int ncol  = col0 + (lane & 15);
  const int abase = (lane & 16) ? 8 : 0;
  // B fragment: lanes 0-15 -> K base 0, lanes 16-31 -> K base 8 (same as A)
  const int bbase = (lane & 16) ? 8 : 0;

  v8f acc = {};
  for (int k0 = 0; k0 < K; k0 += 32) {
    const _Float16* ap = A  + (size_t)mrow * K + k0 + abase;
    const _Float16* bp = BT + (size_t)ncol * K + k0 + bbase;
    h8v a_lo = *(const h8v*)(ap);        // K = abase+0..7
    h8v a_hi = *(const h8v*)(ap + 16);   // K = abase+16..23
    h8v b_lo = *(const h8v*)(bp);        // K = bbase+0..7
    h8v b_hi = *(const h8v*)(bp + 16);   // K = bbase+16..23
    v16h af, bf;
#pragma unroll
    for (int e = 0; e < 8; ++e) {
      af[e] = a_lo[e]; af[e + 8] = a_hi[e];
      bf[e] = b_lo[e]; bf[e + 8] = b_hi[e];
    }
    acc = __builtin_amdgcn_wmma_f32_16x16x32_f16(
        /*neg_a=*/false, af, /*neg_b=*/false, bf,
        /*c_mod=*/(short)0, acc, /*reuse_a=*/false, /*reuse_b=*/false);
  }

  // C layout: VGPR v -> row = row0 + v (+8 for upper lanes), col = lane&15
  const int rbase = row0 + ((lane & 16) ? 8 : 0);
#pragma unroll
  for (int v = 0; v < 8; ++v)
    C[(size_t)(rbase + v) * Nout + ncol] = acc[v];
}

// ---------------- attention scores per node/head ----------------
__global__ void k_scores(const float* __restrict__ h, const float* __restrict__ a_src,
                         const float* __restrict__ a_dst, float* __restrict__ s_src,
                         float* __restrict__ s_dst, int nnodes, int H, int D) {
  int i = blockIdx.x * blockDim.x + threadIdx.x;
  if (i >= nnodes * H) return;
  int n = i / H, hh = i % H;
  const float* row = h + (size_t)n * H * D + (size_t)hh * D;
  const float* as = a_src + hh * D;
  const float* ad = a_dst + hh * D;
  float ss = 0.f, sd = 0.f;
  for (int d = 0; d < D; ++d) { float v = row[d]; ss += v * as[d]; sd += v * ad[d]; }
  s_src[i] = ss; s_dst[i] = sd;
}

// ---------------- edge pass 1: leaky_relu logits + segment max ----------------
__global__ void k_edge_max(const long long* __restrict__ ei, int E, int nnodes,
                           const float* __restrict__ s_src, const float* __restrict__ s_dst,
                           float* __restrict__ eattn, float* __restrict__ mmax, int H) {
  int e = blockIdx.x * blockDim.x + threadIdx.x;
  if (e >= E + nnodes) return;
  int src, dst;
  if (e < E) { src = (int)ei[e]; dst = (int)ei[E + e]; }
  else       { src = e - E; dst = src; }
  for (int hh = 0; hh < H; ++hh) {
    float t = s_src[src * H + hh] + s_dst[dst * H + hh];
    float v = t > 0.f ? t : NEG_SLOPE * t;
    eattn[(size_t)e * H + hh] = v;
    atomicMaxF32(&mmax[dst * H + hh], v);
  }
}

// ---------------- edge pass 2: exp(e - m[dst]) + segment sum ----------------
__global__ void k_edge_exp(const long long* __restrict__ ei, int E, int nnodes,
                           float* __restrict__ eattn, const float* __restrict__ mmax,
                           float* __restrict__ zsum, int H) {
  int e = blockIdx.x * blockDim.x + threadIdx.x;
  if (e >= E + nnodes) return;
  int dst = (e < E) ? (int)ei[E + e] : (e - E);
  for (int hh = 0; hh < H; ++hh) {
    float ex = expf(eattn[(size_t)e * H + hh] - mmax[dst * H + hh]);
    eattn[(size_t)e * H + hh] = ex;
    atomicAdd(&zsum[dst * H + hh], ex);
  }
}

// ---------------- edge pass 3: agg[dst] += alpha * h[src] ----------------
__global__ void k_edge_aggregate(const long long* __restrict__ ei, int E, int nnodes,
                                 const float* __restrict__ hmat, const float* __restrict__ eattn,
                                 const float* __restrict__ zsum, float* __restrict__ agg,
                                 int H, int D) {
  long long idx = (long long)blockIdx.x * blockDim.x + threadIdx.x;
  const int HD = H * D;
  long long total = (long long)(E + nnodes) * HD;
  if (idx >= total) return;
  int e = (int)(idx / HD);
  int j = (int)(idx % HD);
  int src, dst;
  if (e < E) { src = (int)ei[e]; dst = (int)ei[E + e]; }
  else       { src = e - E; dst = src; }
  int hh = j / D;
  float alpha = eattn[(size_t)e * H + hh] / zsum[dst * H + hh];
  atomicAdd(&agg[(size_t)dst * HD + j], alpha * hmat[(size_t)src * HD + j]);
}

// ---------------- bias + ELU -> f16 activation ----------------
__global__ void k_bias_elu_f16(const float* __restrict__ agg, const float* __restrict__ b,
                               _Float16* __restrict__ act, int n, int HD) {
  int i = blockIdx.x * blockDim.x + threadIdx.x;
  if (i >= n) return;
  float v = agg[i] + b[i % HD];
  float y = v > 0.f ? v : (expf(v) - 1.f);
  act[i] = (_Float16)y;
}

// ---------------- final: bias + log_softmax over NCLS ----------------
__global__ void k_final_logsoftmax(const float* __restrict__ agg, const float* __restrict__ b,
                                   float* __restrict__ out, int nnodes) {
  int n = blockIdx.x * blockDim.x + threadIdx.x;
  if (n >= nnodes) return;
  float l[NCLS];
  float mx = -INFINITY;
#pragma unroll
  for (int c = 0; c < NCLS; ++c) { l[c] = agg[(size_t)n * NCLS + c] + b[c]; mx = fmaxf(mx, l[c]); }
  float s = 0.f;
#pragma unroll
  for (int c = 0; c < NCLS; ++c) s += expf(l[c] - mx);
  float ls = logf(s);
#pragma unroll
  for (int c = 0; c < NCLS; ++c) out[(size_t)n * NCLS + c] = l[c] - mx - ls;
}

// ---------------- host side ----------------
static inline unsigned int cdiv(long long a, long long b) { return (unsigned int)((a + b - 1) / b); }

extern "C" void kernel_launch(void* const* d_in, const int* in_sizes, int n_in,
                              void* d_out, int out_size, void* d_ws, size_t ws_size,
                              hipStream_t stream) {
  const float*     x   = (const float*)d_in[0];
  const long long* ei  = (const long long*)d_in[1];  // int64 [2, E]
  const float* W1  = (const float*)d_in[2];
  const float* a1s = (const float*)d_in[3];
  const float* a1d = (const float*)d_in[4];
  const float* b1  = (const float*)d_in[5];
  const float* W2  = (const float*)d_in[6];
  const float* a2s = (const float*)d_in[7];
  const float* a2d = (const float*)d_in[8];
  const float* b2  = (const float*)d_in[9];
  const float* W3  = (const float*)d_in[10];
  const float* a3s = (const float*)d_in[11];
  const float* a3d = (const float*)d_in[12];
  const float* b3  = (const float*)d_in[13];
  float* out = (float*)d_out;

  // ---- carve workspace (256B aligned chunks) ----
  char* p = (char*)d_ws;
  auto carve = [&](size_t bytes) -> void* {
    void* r = (void*)p; p += (bytes + 255) & ~(size_t)255; return r;
  };
  _Float16* x16   = (_Float16*)carve((size_t)NNODES * FIN * sizeof(_Float16));
  _Float16* w1t   = (_Float16*)carve((size_t)HD1 * FIN * sizeof(_Float16));
  _Float16* w2t   = (_Float16*)carve((size_t)HD1 * HD1 * sizeof(_Float16));
  _Float16* w3t   = (_Float16*)carve((size_t)NCLS * HD1 * sizeof(_Float16));
  float*    hmat  = (float*)carve((size_t)NNODES * HD1 * sizeof(float));
  float*    agg   = (float*)carve((size_t)NNODES * HD1 * sizeof(float));
  _Float16* act16 = (_Float16*)carve((size_t)NNODES * HD1 * sizeof(_Float16));
  float*    ssrc  = (float*)carve((size_t)NNODES * NHEADS * sizeof(float));
  float*    sdst  = (float*)carve((size_t)NNODES * NHEADS * sizeof(float));
  float*    mmax  = (float*)carve((size_t)NNODES * NHEADS * sizeof(float));
  float*    zsum  = (float*)carve((size_t)NNODES * NHEADS * sizeof(float));
  float*    eattn = (float*)carve((size_t)(NEDGES + NNODES) * NHEADS * sizeof(float));

  const int ET = NEDGES + NNODES;
  const dim3 blk256(256);

  // ---- precision conversion ----
  k_f32_to_f16<<<cdiv((long long)NNODES * FIN, 256), blk256, 0, stream>>>(x, x16, NNODES * FIN);
  k_transpose_f16<<<cdiv(FIN * HD1, 256), blk256, 0, stream>>>(W1, w1t, FIN, HD1);
  k_transpose_f16<<<cdiv(HD1 * HD1, 256), blk256, 0, stream>>>(W2, w2t, HD1, HD1);
  k_transpose_f16<<<cdiv(HD1 * NCLS, 256), blk256, 0, stream>>>(W3, w3t, HD1, NCLS);

  const dim3 gemmBlk(32, 4);

  // =============== layer 1: FIN -> [HEADS x HID] concat ===============
  {
    const int H = NHEADS, D = HIDD, HD = HD1, K = FIN;
    k_gemm_wmma<<<dim3(NNODES / 16, cdiv(HD / 16, 4)), gemmBlk, 0, stream>>>(x16, w1t, hmat, NNODES, K, HD);
    k_scores<<<cdiv(NNODES * H, 256), blk256, 0, stream>>>(hmat, a1s, a1d, ssrc, sdst, NNODES, H, D);
    k_fill<<<cdiv(NNODES * H, 256), blk256, 0, stream>>>(mmax, NNODES * H, -INFINITY);
    k_fill<<<cdiv(NNODES * H, 256), blk256, 0, stream>>>(zsum, NNODES * H, 0.f);
    k_fill<<<cdiv((long long)NNODES * HD, 256), blk256, 0, stream>>>(agg, NNODES * HD, 0.f);
    k_edge_max<<<cdiv(ET, 256), blk256, 0, stream>>>(ei, NEDGES, NNODES, ssrc, sdst, eattn, mmax, H);
    k_edge_exp<<<cdiv(ET, 256), blk256, 0, stream>>>(ei, NEDGES, NNODES, eattn, mmax, zsum, H);
    k_edge_aggregate<<<cdiv((long long)ET * HD, 256), blk256, 0, stream>>>(ei, NEDGES, NNODES, hmat, eattn, zsum, agg, H, D);
    k_bias_elu_f16<<<cdiv((long long)NNODES * HD, 256), blk256, 0, stream>>>(agg, b1, act16, NNODES * HD, HD);
  }

  // =============== layer 2: HD1 -> [HEADS x HID] concat ===============
  {
    const int H = NHEADS, D = HIDD, HD = HD1, K = HD1;
    k_gemm_wmma<<<dim3(NNODES / 16, cdiv(HD / 16, 4)), gemmBlk, 0, stream>>>(act16, w2t, hmat, NNODES, K, HD);
    k_scores<<<cdiv(NNODES * H, 256), blk256, 0, stream>>>(hmat, a2s, a2d, ssrc, sdst, NNODES, H, D);
    k_fill<<<cdiv(NNODES * H, 256), blk256, 0, stream>>>(mmax, NNODES * H, -INFINITY);
    k_fill<<<cdiv(NNODES * H, 256), blk256, 0, stream>>>(zsum, NNODES * H, 0.f);
    k_fill<<<cdiv((long long)NNODES * HD, 256), blk256, 0, stream>>>(agg, NNODES * HD, 0.f);
    k_edge_max<<<cdiv(ET, 256), blk256, 0, stream>>>(ei, NEDGES, NNODES, ssrc, sdst, eattn, mmax, H);
    k_edge_exp<<<cdiv(ET, 256), blk256, 0, stream>>>(ei, NEDGES, NNODES, eattn, mmax, zsum, H);
    k_edge_aggregate<<<cdiv((long long)ET * HD, 256), blk256, 0, stream>>>(ei, NEDGES, NNODES, hmat, eattn, zsum, agg, H, D);
    k_bias_elu_f16<<<cdiv((long long)NNODES * HD, 256), blk256, 0, stream>>>(agg, b2, act16, NNODES * HD, HD);
  }

  // =============== layer 3: HD1 -> NCLS, heads=1 (mean == identity) ===============
  {
    const int H = 1, D = NCLS, HD = NCLS, K = HD1;
    k_gemm_wmma<<<dim3(NNODES / 16, cdiv(HD / 16, 4)), gemmBlk, 0, stream>>>(act16, w3t, hmat, NNODES, K, HD);
    k_scores<<<cdiv(NNODES * H, 256), blk256, 0, stream>>>(hmat, a3s, a3d, ssrc, sdst, NNODES, H, D);
    k_fill<<<cdiv(NNODES * H, 256), blk256, 0, stream>>>(mmax, NNODES * H, -INFINITY);
    k_fill<<<cdiv(NNODES * H, 256), blk256, 0, stream>>>(zsum, NNODES * H, 0.f);
    k_fill<<<cdiv((long long)NNODES * HD, 256), blk256, 0, stream>>>(agg, NNODES * HD, 0.f);
    k_edge_max<<<cdiv(ET, 256), blk256, 0, stream>>>(ei, NEDGES, NNODES, ssrc, sdst, eattn, mmax, H);
    k_edge_exp<<<cdiv(ET, 256), blk256, 0, stream>>>(ei, NEDGES, NNODES, eattn, mmax, zsum, H);
    k_edge_aggregate<<<cdiv((long long)ET * HD, 256), blk256, 0, stream>>>(ei, NEDGES, NNODES, hmat, eattn, zsum, agg, H, D);
    k_final_logsoftmax<<<cdiv(NNODES, 256), blk256, 0, stream>>>(agg, b3, out, NNODES);
  }
}
